// GraphAttentionNetwork_17179869621
// MI455X (gfx1250) — matrix operand order repaired and checked by reading, not compiled, not run on hardware
//
#include <hip/hip_runtime.h>
#include <cstdint>
#include <cstddef>

// Problem constants (match reference)
#define BB   16
#define NN   512
#define DD   512     // D_IN == D_OUT == H*F
#define HH   8
#define FF   64
#define KSEL 256     // top-k = N/2
#define NEG_SLOPE 0.1f

typedef __attribute__((ext_vector_type(16))) __bf16   v16bf;
typedef __attribute__((ext_vector_type(8)))  __bf16   v8bf;
typedef __attribute__((ext_vector_type(8)))  float    v8f;
typedef __attribute__((ext_vector_type(4)))  uint32_t v4u;
typedef __attribute__((ext_vector_type(8)))  uint32_t v8u;

__device__ __forceinline__ float leaky_f(float x) { return x >= 0.f ? x : NEG_SLOPE * x; }

// LDS byte offset of a __shared__ object (low 32 bits of the flat address).
__device__ __forceinline__ uint32_t lds_off_u32(const void* p) {
  return (uint32_t)(uintptr_t)p;
}

// ---- CDNA5 async global->LDS copy (ASYNCcnt), per-lane 16B ----
__device__ __forceinline__ void async_copy_b128(uint32_t ldsoff, const void* gptr) {
  asm volatile("global_load_async_to_lds_b128 %0, %1, off"
               :: "v"(ldsoff), "v"(gptr)
               : "memory");
}
__device__ __forceinline__ void wait_async(int allow) {  // ASYNCcnt <= allow
  if (allow == 0)      asm volatile("s_wait_asynccnt 0" ::: "memory");
  else                 asm volatile("s_wait_asynccnt 1" ::: "memory");
}

// ---- CDNA5 Tensor Data Mover: 2D strided tile -> LDS, one instruction ----
// Builds D# group0/group1 per ISA 08_async_tensor §8.3/8.4, 2-SGPR-group form
// (VADDR2/3 omitted => tensors up to 2D). Issue from one wave; TENSORcnt tracked.
__device__ __forceinline__ void tdm_load_2d(uint32_t lds_addr, const void* gaddr,
                                            uint32_t tensor_d0, uint32_t tensor_d1,
                                            uint32_t tile_d0, uint32_t tile_d1,
                                            uint32_t stride_d0) {
  uint64_t ga = (uint64_t)(uintptr_t)gaddr;
  v4u g0;
  g0[0] = 1u;                                                  // count=1, user D#
  g0[1] = lds_addr;                                            // lds_addr (bytes)
  g0[2] = (uint32_t)ga;                                        // global_addr[31:0]
  g0[3] = (uint32_t)((ga >> 32) & 0x01FFFFFFu) | (2u << 30);   // addr[56:32] | type=2
  v8u g1;
  g1[0] = (1u << 16);                                          // wg_mask=0, data_size=1 (2B)
  g1[1] = (tensor_d0 & 0xFFFFu) << 16;                         // tensor_dim0 lo16
  g1[2] = (tensor_d0 >> 16) | ((tensor_d1 & 0xFFFFu) << 16);   // d0 hi | d1 lo
  g1[3] = (tensor_d1 >> 16) | (tile_d0 << 16);                 // d1 hi | tile_dim0
  g1[4] = tile_d1;                                             // tile_dim1 (tile_dim2=0)
  g1[5] = stride_d0;                                           // tensor_dim0_stride[31:0]
  g1[6] = 0u;
  g1[7] = 0u;
  asm volatile("tensor_load_to_lds %0, %1" :: "s"(g0), "s"(g1) : "memory");
}

// ---------------------------------------------------------------------------
// Converters (layer-1 input + weight pre-transpose only; everything else fused)
// ---------------------------------------------------------------------------
__global__ __launch_bounds__(256) void conv_bf16_kernel(const float* __restrict__ x,
                                                        __bf16* __restrict__ y, int n) {
  for (int i = blockIdx.x * blockDim.x + threadIdx.x; i < n; i += gridDim.x * blockDim.x)
    y[i] = (__bf16)x[i];
}

// Wt[n][k] = W[k][n]  (bf16), 512x512
__global__ __launch_bounds__(256) void conv_wT_kernel(const float* __restrict__ W,
                                                      __bf16* __restrict__ Wt) {
  int t = blockIdx.x * blockDim.x + threadIdx.x;
  if (t >= DD * DD) return;
  int n = t >> 9, k = t & 511;
  Wt[(size_t)n * DD + k] = (__bf16)W[(size_t)k * DD + n];
}

// ---------------------------------------------------------------------------
// Feature GEMM with fused epilogue.
//   in : A_bf [8192,512] bf16, Wt [n][k] bf16 pre-transposed, a[2F]
//   out: gT[b][h][f][n] bf16 (per-head transposed), el/er[b,n,h] f32
// Tile 128(M) x 64(N) (block covers exactly one head), K-step 32, double-buffered.
// A tile staged by TDM (tensor_load_to_lds), B tile by per-lane async copies.
// All four B fragments preloaded so the four WMMAs issue back-to-back.
// ---------------------------------------------------------------------------
__global__ __launch_bounds__(256) void gemm_bf16_kernel(const __bf16* __restrict__ A,
                                                        const __bf16* __restrict__ Wt,
                                                        const float* __restrict__ avec,
                                                        __bf16* __restrict__ gT,
                                                        float* __restrict__ el,
                                                        float* __restrict__ er) {
  __shared__ alignas(128) __bf16 As[2][128][32];   // [m][k]
  __shared__ alignas(128) __bf16 Bs[2][64][32];    // [n][k] (transposed)

  const int tid  = threadIdx.x;
  const int wave = tid >> 5;
  const int lane = tid & 31;
  const int row0 = blockIdx.x * 128;
  const int col0 = blockIdx.y * 64;
  const int head = blockIdx.y;                     // TN==F so block == one head

  v8f acc[4];
#pragma unroll
  for (int t = 0; t < 4; ++t)
#pragma unroll
    for (int i = 0; i < 8; ++i) acc[t][i] = 0.f;

  const int arow = 16 * wave + (lane & 15);
  const int ak0  = (lane < 16) ? 0 : 16;
  const int bk0  = (lane < 16) ? 0 : 16;
  const int brow = tid >> 2, bseg = tid & 3;

  // ---- prime stage for buffer 0 ----
  async_copy_b128(lds_off_u32(&Bs[0][brow][bseg * 8]),
                  Wt + (size_t)(col0 + brow) * DD + bseg * 8);
  if (wave == 0)
    tdm_load_2d(lds_off_u32(&As[0][0][0]), A + (size_t)row0 * DD,
                DD, BB * NN, 32, 128, DD);

  for (int k0 = 0; k0 < DD; k0 += 32) {
    const int p = (k0 >> 5) & 1;
    const bool next = (k0 + 32) < DD;
    if (next) {
      async_copy_b128(lds_off_u32(&Bs[p ^ 1][brow][bseg * 8]),
                      Wt + (size_t)(col0 + brow) * DD + k0 + 32 + bseg * 8);
      if (wave == 0)
        tdm_load_2d(lds_off_u32(&As[p ^ 1][0][0]), A + (size_t)row0 * DD + k0 + 32,
                    DD, BB * NN, 32, 128, DD);
      __builtin_prefetch(Wt + (size_t)(col0 + brow) * DD + k0 + 64, 0, 1);
    }
    wait_async(next ? 1 : 0);
    if (wave == 0) {
      if (next) __builtin_amdgcn_s_wait_tensorcnt(1);
      else      __builtin_amdgcn_s_wait_tensorcnt(0);
    }
    __syncthreads();

    // Preload A fragment and ALL four B fragments, then issue the WMMA chain
    // back-to-back (single DS wait, keeps the matrix pipe saturated).
    v8bf alo = *(const v8bf*)&As[p][arow][ak0];
    v8bf ahi = *(const v8bf*)&As[p][arow][ak0 + 8];
    v16bf afrag;
#pragma unroll
    for (int i = 0; i < 8; ++i) { afrag[i] = alo[i]; afrag[i + 8] = ahi[i]; }

    v16bf bfrag[4];
#pragma unroll
    for (int t = 0; t < 4; ++t)
      bfrag[t] = *(const v16bf*)&Bs[p][16 * t + (lane & 15)][bk0];

#pragma unroll
    for (int t = 0; t < 4; ++t)
      acc[t] = __builtin_amdgcn_wmma_f32_16x16x32_bf16(
          false, afrag, false, bfrag[t], (short)0, acc[t], false, false);
    __syncthreads();
  }

  // ---- fused epilogue ----
  const int rbase = (lane >= 16) ? 8 : 0;
  const int mbase = row0 + 16 * wave + rbase;      // global row of acc[.][0]
  const int bidx  = mbase >> 9;                    // batch (128 | 512 so no straddle)
  const int nidx  = mbase & 511;                   // node within batch

  // (a) gT[b][h][f][n]: 8 consecutive n per lane -> one packed 16B store per subtile
#pragma unroll
  for (int t = 0; t < 4; ++t) {
    const int f = 16 * t + (lane & 15);
    v8bf pk;
#pragma unroll
    for (int r = 0; r < 8; ++r) pk[r] = (__bf16)acc[t][r];
    *(v8bf*)(gT + ((size_t)(bidx * HH + head) * FF + f) * NN + nidx) = pk;
  }

  // (b) el/er via 16-lane xor-shuffle reduction (halves hold distinct rows)
  float aL[4], aR[4];
#pragma unroll
  for (int t = 0; t < 4; ++t) {
    aL[t] = avec[16 * t + (lane & 15)];
    aR[t] = avec[FF + 16 * t + (lane & 15)];
  }
#pragma unroll
  for (int r = 0; r < 8; ++r) {
    float sl = 0.f, sr = 0.f;
#pragma unroll
    for (int t = 0; t < 4; ++t) { sl += acc[t][r] * aL[t]; sr += acc[t][r] * aR[t]; }
#pragma unroll
    for (int mask = 8; mask >= 1; mask >>= 1) {
      sl += __shfl_xor(sl, mask, 32);
      sr += __shfl_xor(sr, mask, 32);
    }
    if ((lane & 15) == 0) {
      const int node = mbase + r;                  // == b*NN + n
      el[(size_t)node * HH + head] = sl;
      er[(size_t)node * HH + head] = sr;
    }
  }
}

// ---------------------------------------------------------------------------
// Flash-style masked attention + aggregation, one (b, h, 16-row block) per block.
// 128 threads = 4 waves; wave w owns output cols [16w, 16w+16).
// gT chunks staged by TDM, double-buffered. Softmax exp distributed over all
// threads with ds_add_f32 row sums. Writes f32 h and bf16 h (next layer input).
// ---------------------------------------------------------------------------
__global__ __launch_bounds__(128) void gat_attn_kernel(const __bf16* __restrict__ gT,
                                                       const float* __restrict__ el,
                                                       const float* __restrict__ er,
                                                       const float* __restrict__ adj,
                                                       float* __restrict__ hout,
                                                       __bf16* __restrict__ hbf) {
  __shared__ alignas(128) __bf16 gTs[2][64][32];  // [f][j] transposed B tiles
  __shared__ alignas(128) __bf16 probs[16][32];   // A tile (softmax numerators)
  __shared__ float es[16][32];
  __shared__ float mrow[16], lrow[16], srow[16], psum[16], elrow[16];

  const int b  = blockIdx.z;
  const int h  = blockIdx.y;
  const int i0 = blockIdx.x * 16;
  const int tid = threadIdx.x, wave = tid >> 5, lane = tid & 31;

  v8f acc;
#pragma unroll
  for (int i = 0; i < 8; ++i) acc[i] = 0.f;

  if (tid < 16) {
    mrow[tid] = -1e30f;
    lrow[tid] = 0.f;
    elrow[tid] = el[((size_t)b * NN + i0 + tid) * HH + h];
  }

  const __bf16* gTh = gT + ((size_t)(b * HH + h) * FF) * NN;  // [f][n]
  const int arow = lane & 15;
  const int ak0  = (lane < 16) ? 0 : 16;
  const int bk0  = (lane < 16) ? 0 : 16;
  const int fcol = 16 * wave + (lane & 15);
  const int rbase = (lane >= 16) ? 8 : 0;

  if (wave == 0)                                   // prime buffer 0
    tdm_load_2d(lds_off_u32(&gTs[0][0][0]), gTh, NN, FF, 32, FF, NN);
  __syncthreads();

  for (int j0 = 0; j0 < NN; j0 += 32) {
    const int p = (j0 >> 5) & 1;
    const bool next = (j0 + 32) < NN;
    if (next && wave == 0)
      tdm_load_2d(lds_off_u32(&gTs[p ^ 1][0][0]), gTh + j0 + 32, NN, FF, 32, FF, NN);

    // masked leaky logits for this 16x32 tile (4 per thread)
    for (int q = tid; q < 512; q += 128) {
      int r = q >> 5, c = q & 31;
      int i = i0 + r, j = j0 + c;
      float aadj = adj[((size_t)b * NN + i) * NN + j] + ((i == j) ? 1.f : 0.f);
      float x = elrow[r] + er[((size_t)b * NN + j) * HH + h];
      es[r][c] = (aadj == 0.f) ? -1000.f : leaky_f(x);
    }
    if (tid < 16) psum[tid] = 0.f;
    __syncthreads();                               // B1: es + psum ready

    if (tid < 16) {                                // per-row running max
      int r = tid;
      float mo = mrow[r], mn = mo;
#pragma unroll
      for (int c = 0; c < 32; ++c) mn = fmaxf(mn, es[r][c]);
      srow[r] = __expf(mo - mn);
      mrow[r] = mn;
    }
    __syncthreads();                               // B2: mrow/srow ready

    for (int q = tid; q < 512; q += 128) {         // distributed exp + row sums
      int r = q >> 5, c = q & 31;
      float pe = __expf(es[r][c] - mrow[r]);
      probs[r][c] = (__bf16)pe;
      atomicAdd(&psum[r], pe);                     // ds_add_f32
    }
    if (wave == 0) {
      if (next) __builtin_amdgcn_s_wait_tensorcnt(1);
      else      __builtin_amdgcn_s_wait_tensorcnt(0);
    }
    __syncthreads();                               // B3: probs/psum/gTs[p] ready

    if (tid < 16) lrow[tid] = lrow[tid] * srow[tid] + psum[tid];

#pragma unroll
    for (int r = 0; r < 8; ++r) acc[r] *= srow[rbase + r];

    v8bf alo = *(const v8bf*)&probs[arow][ak0];
    v8bf ahi = *(const v8bf*)&probs[arow][ak0 + 8];
    v16bf afrag;
#pragma unroll
    for (int i = 0; i < 8; ++i) { afrag[i] = alo[i]; afrag[i + 8] = ahi[i]; }
    v16bf bfrag = *(const v16bf*)&gTs[p][fcol][bk0];
    acc = __builtin_amdgcn_wmma_f32_16x16x32_bf16(
        false, afrag, false, bfrag, (short)0, acc, false, false);
    __syncthreads();                               // B4: LDS reuse
  }

  // epilogue: normalize, trailing leaky, write f32 h and bf16 h
#pragma unroll
  for (int r = 0; r < 8; ++r) {
    int row = i0 + rbase + r;
    float v = leaky_f(acc[r] / lrow[rbase + r]);
    size_t o = ((size_t)b * NN + row) * DD + h * FF + fcol;
    hout[o] = v;
    hbf[o]  = (__bf16)v;
  }
}

// ---------------------------------------------------------------------------
// Pool score: scores[b,n] = h[b,n,:].Wp + bp
// ---------------------------------------------------------------------------
__global__ __launch_bounds__(256) void score_kernel(const float* __restrict__ hbuf,
                                                    const float* __restrict__ Wp,
                                                    const float* __restrict__ bp,
                                                    float* __restrict__ sc) {
  int t = blockIdx.x * blockDim.x + threadIdx.x;
  if (t >= BB * NN) return;
  const float* hp = hbuf + (size_t)t * DD;
  float acc = bp[0];
#pragma unroll 8
  for (int d = 0; d < DD; ++d) acc += hp[d] * Wp[d];
  sc[t] = acc;
}

// ---------------------------------------------------------------------------
// Per-batch top-256 of 512 via LDS bitonic sort (desc score, asc index ties),
// then gather selected rows of h into d_out.
// ---------------------------------------------------------------------------
__global__ __launch_bounds__(512) void topk_gather_kernel(const float* __restrict__ sc,
                                                          const float* __restrict__ hbuf,
                                                          float* __restrict__ out) {
  __shared__ float key[NN];
  __shared__ int   kid[NN];
  const int b = blockIdx.x, t = threadIdx.x;
  key[t] = sc[(size_t)b * NN + t];
  kid[t] = t;
  __syncthreads();

  for (int size = 2; size <= NN; size <<= 1) {
    for (int stride = size >> 1; stride > 0; stride >>= 1) {
      if (t < NN / 2) {
        int i = 2 * t - (t & (stride - 1));
        int j = i + stride;
        bool dirDesc = ((i & size) == 0);
        float ki = key[i], kj = key[j];
        int ii = kid[i], ij = kid[j];
        bool iFirst = (ki > kj) || (ki == kj && ii < ij);
        if (dirDesc ? !iFirst : iFirst) {
          key[i] = kj; key[j] = ki;
          kid[i] = ij; kid[j] = ii;
        }
      }
      __syncthreads();
    }
  }

  for (int q = t; q < KSEL * DD; q += NN) {
    int kk = q >> 9, d = q & 511;
    out[((size_t)b * KSEL + kk) * DD + d] = hbuf[((size_t)b * NN + kid[kk]) * DD + d];
  }
}

// ---------------------------------------------------------------------------
// Host launcher
// ---------------------------------------------------------------------------
extern "C" void kernel_launch(void* const* d_in, const int* in_sizes, int n_in,
                              void* d_out, int out_size, void* d_ws, size_t ws_size,
                              hipStream_t stream) {
  (void)in_sizes; (void)n_in; (void)out_size; (void)ws_size;
  const float* x   = (const float*)d_in[0];
  const float* adj = (const float*)d_in[1];
  const float* W1  = (const float*)d_in[2];
  const float* a1  = (const float*)d_in[3];
  const float* W2  = (const float*)d_in[4];
  const float* a2  = (const float*)d_in[5];
  const float* Wp  = (const float*)d_in[6];
  const float* bp  = (const float*)d_in[7];
  float* out = (float*)d_out;

  char* ws = (char*)d_ws;
  size_t off = 0;
  auto take = [&](size_t bytes) -> void* {
    void* p = ws + off;
    off = (off + bytes + 255) & ~(size_t)255;
    return p;
  };
  float*  h1   = (float*)take((size_t)BB * NN * DD * 4);   // layer-1 f32 h
  float*  h2   = (float*)take((size_t)BB * NN * DD * 4);   // layer-2 f32 h
  __bf16* Abf  = (__bf16*)take((size_t)BB * NN * DD * 2);  // bf16 GEMM input (reused)
  __bf16* WtT  = (__bf16*)take((size_t)DD * DD * 2);
  __bf16* gT   = (__bf16*)take((size_t)BB * NN * DD * 2);
  __bf16* hbf2 = (__bf16*)take((size_t)BB * NN * DD * 2);  // scratch bf16 (layer-2)
  float*  el   = (float*)take((size_t)BB * NN * HH * 4);
  float*  er   = (float*)take((size_t)BB * NN * HH * 4);
  float*  sc   = (float*)take((size_t)BB * NN * 4);

  const dim3 gemmGrid(BB * NN / 128, DD / 64);
  const dim3 attnGrid(NN / 16, HH, BB);

  // ---- Layer 1 ----
  conv_bf16_kernel<<<4096, 256, 0, stream>>>(x, Abf, BB * NN * DD);
  conv_wT_kernel<<<(DD * DD) / 256, 256, 0, stream>>>(W1, WtT);
  gemm_bf16_kernel<<<gemmGrid, 256, 0, stream>>>(Abf, WtT, a1, gT, el, er);
  gat_attn_kernel<<<attnGrid, 128, 0, stream>>>(gT, el, er, adj, h1, Abf);
  // (attention wrote bf16 h1 into Abf -> layer-2 GEMM input, no conv pass)

  // ---- Layer 2 ----
  conv_wT_kernel<<<(DD * DD) / 256, 256, 0, stream>>>(W2, WtT);
  gemm_bf16_kernel<<<gemmGrid, 256, 0, stream>>>(Abf, WtT, a2, gT, el, er);
  gat_attn_kernel<<<attnGrid, 128, 0, stream>>>(gT, el, er, adj, h2, hbf2);

  // ---- SagPool ----
  score_kernel<<<(BB * NN) / 256, 256, 0, stream>>>(h2, Wp, bp, sc);
  topk_gather_kernel<<<BB, 512, 0, stream>>>(sc, h2, out);
}
